// BehaviorTransformer_35794257444911
// MI455X (gfx1250) — matrix operand order repaired and checked by reading, not compiled, not run on hardware
//
#include <hip/hip_runtime.h>
#include <hip/hip_bf16.h>
#include <math.h>

typedef __attribute__((ext_vector_type(16))) __bf16 v16bf;
typedef __attribute__((ext_vector_type(8)))  __bf16 v8bf;
typedef __attribute__((ext_vector_type(8)))  float  v8f;

// ---- problem constants ----
constexpr int cB   = 2048;
constexpr int cS   = 60;
constexpr int cIN  = 211;
constexpr int cINP = 224;   // padded K for input projection
constexpr int cD   = 192;
constexpr int cH   = 6;
constexpr int cL   = 3;
constexpr int cDFF = 768;
constexpr int cNC  = 35;
constexpr int cNCP = 64;    // padded N for classifier head
constexpr int cDH  = 96;    // classifier hidden (real)
constexpr int cDHP = 128;   // classifier hidden padded to N-tile multiple
constexpr int cM   = cB * cS; // 122880 tokens, multiple of 128

__device__ __forceinline__ float gelu_exact(float x) {
    return 0.5f * x * (1.0f + erff(x * 0.70710678118654752f));
}

// ---------------------------------------------------------------------------
// Weight transpose + fp32->bf16 convert:  W[K][N] (fp32) -> WT[Npad][Kpad] (bf16)
// ---------------------------------------------------------------------------
__global__ __launch_bounds__(256)
void convT_kernel(const float* __restrict__ W, __bf16* __restrict__ WT,
                  int K, int Kpad, int N, int Npad) {
    int i = blockIdx.x * 256 + threadIdx.x;
    if (i >= Npad * Kpad) return;
    int n = i / Kpad;
    int k = i % Kpad;
    float v = (k < K && n < N) ? W[(size_t)k * N + n] : 0.0f;
    WT[i] = (__bf16)v;
}

// x[M][211] fp32 -> xbf[M][224] bf16 (zero padded)
__global__ __launch_bounds__(256)
void padx_kernel(const float* __restrict__ X, __bf16* __restrict__ Xb) {
    size_t i = (size_t)blockIdx.x * 256 + threadIdx.x;
    if (i >= (size_t)cM * cINP) return;
    size_t m = i / cINP;
    int k = (int)(i % cINP);
    float v = (k < cIN) ? X[m * cIN + k] : 0.0f;
    Xb[i] = (__bf16)v;
}

// ---------------------------------------------------------------------------
// WMMA GEMM:  out = epi( A[M][KK]bf16 @ WT[N][KK]bf16^T + bias )
// block = 256 threads = 8 waves; block tile 128(M) x 64(N).
// Each wave owns a 32x32 tile = 2x2 WMMA tiles (4 v_wmma per K-step),
// reusing each A/B fragment twice -> 32 B/lane of VMEM per WMMA.
// KK compile-time -> fully unrolled, clause-batched loads.
// EPI: 0 store bf16 (qkv)      1 +pe, store f32+bf16 (input proj)
//      2 +resid, store f32     3 GELU, store bf16    4 store f32, col<nReal
// ---------------------------------------------------------------------------
template <int EPI, int KK>
__global__ __launch_bounds__(256)
void gemm_wmma_bf16(const __bf16* __restrict__ A, int lda,
                    const __bf16* __restrict__ Bt, int ldb,
                    const float* __restrict__ bias,
                    const float* __restrict__ extra,  // pe (EPI=1) or residual (EPI=2)
                    float* __restrict__ outF, __bf16* __restrict__ outB,
                    int ldout, int nReal) {
    const int lane = threadIdx.x & 31;
    const int wave = threadIdx.x >> 5;
    const int l16  = lane & 15;
    const int half = lane >> 4;
    const int mt = blockIdx.x * 128 + (wave & 3) * 32;   // wave's 32-row strip
    const int nt = blockIdx.y * 64 + (wave >> 2) * 32;   // wave's 32-col strip

    const __bf16* arow0 = A  + (size_t)(mt + l16) * lda;
    const __bf16* arow1 = A  + (size_t)(mt + 16 + l16) * lda;
    const __bf16* brow0 = Bt + (size_t)(nt + l16) * ldb;
    const __bf16* brow1 = Bt + (size_t)(nt + 16 + l16) * ldb;

    v8f acc00 = {}, acc01 = {}, acc10 = {}, acc11 = {};
    for (int k0 = 0; k0 < KK; k0 += 32) {
        // A fragment (contiguous order, matching B): lane half h holds
        // K = {16h..16h+15} of row l16
        v8bf a0lo = *(const v8bf*)(arow0 + k0 + 16 * half);
        v8bf a0hi = *(const v8bf*)(arow0 + k0 + 16 * half + 8);
        v8bf a1lo = *(const v8bf*)(arow1 + k0 + 16 * half);
        v8bf a1hi = *(const v8bf*)(arow1 + k0 + 16 * half + 8);
        // B fragment: lanes 0-15 hold K=0..15, lanes 16-31 K=16..31 of col l16
        v16bf b0 = *(const v16bf*)(brow0 + k0 + 16 * half);
        v16bf b1 = *(const v16bf*)(brow1 + k0 + 16 * half);
        v16bf a0, a1;
#pragma unroll
        for (int i = 0; i < 8; ++i) {
            a0[i] = a0lo[i]; a0[i + 8] = a0hi[i];
            a1[i] = a1lo[i]; a1[i + 8] = a1hi[i];
        }
        acc00 = __builtin_amdgcn_wmma_f32_16x16x32_bf16(false, a0, false, b0,
                                                        (short)0, acc00, false, false);
        acc01 = __builtin_amdgcn_wmma_f32_16x16x32_bf16(false, a0, false, b1,
                                                        (short)0, acc01, false, false);
        acc10 = __builtin_amdgcn_wmma_f32_16x16x32_bf16(false, a1, false, b0,
                                                        (short)0, acc10, false, false);
        acc11 = __builtin_amdgcn_wmma_f32_16x16x32_bf16(false, a1, false, b1,
                                                        (short)0, acc11, false, false);
        if (k0 + 32 < KK) {
            __builtin_prefetch(arow0 + k0 + 32, 0, 1);
            __builtin_prefetch(arow1 + k0 + 32, 0, 1);
        }
    }

    const v8f* accs[2][2] = {{&acc00, &acc01}, {&acc10, &acc11}};
#pragma unroll
    for (int mi = 0; mi < 2; ++mi) {
#pragma unroll
        for (int ni = 0; ni < 2; ++ni) {
            const v8f a = *accs[mi][ni];
            const int n = nt + ni * 16 + l16;
            const float bval = (n < nReal) ? bias[n] : 0.0f;
#pragma unroll
            for (int v = 0; v < 8; ++v) {
                const int m = mt + mi * 16 + 8 * half + v;  // C layout: VGPR v = row 8*half+v
                float val = a[v];
                if constexpr (EPI == 4) {
                    if (n < nReal) outF[(size_t)m * ldout + n] = val + bval;
                } else {
                    val += bval;
                    if constexpr (EPI == 0) {
                        outB[(size_t)m * ldout + n] = (__bf16)val;
                    } else if constexpr (EPI == 1) {
                        val += extra[(size_t)(m % cS) * cD + n];
                        outF[(size_t)m * ldout + n] = val;
                        outB[(size_t)m * ldout + n] = (__bf16)val;
                    } else if constexpr (EPI == 2) {
                        outF[(size_t)m * ldout + n] = val + extra[(size_t)m * ldout + n];
                    } else if constexpr (EPI == 3) {
                        outB[(size_t)m * ldout + n] = (__bf16)gelu_exact(val);
                    }
                }
            }
        }
    }
}

// ---------------------------------------------------------------------------
// LayerNorm: one wave per token (192 = 32 lanes x 6); writes fp32 + bf16
// ---------------------------------------------------------------------------
__global__ __launch_bounds__(256)
void ln_kernel(const float* __restrict__ in, const float* __restrict__ g,
               const float* __restrict__ bta, float* __restrict__ outF,
               __bf16* __restrict__ outB) {
    const int lane = threadIdx.x & 31;
    const int wave = threadIdx.x >> 5;
    const size_t tok = (size_t)blockIdx.x * 8 + wave;
    const float* row = in + tok * cD;
    float x[6], s = 0.0f, s2 = 0.0f;
#pragma unroll
    for (int j = 0; j < 6; ++j) {
        x[j] = row[lane + 32 * j];
        s  += x[j];
        s2 += x[j] * x[j];
    }
#pragma unroll
    for (int m = 1; m < 32; m <<= 1) {
        s  += __shfl_xor(s,  m, 32);
        s2 += __shfl_xor(s2, m, 32);
    }
    const float mean = s * (1.0f / cD);
    const float var  = s2 * (1.0f / cD) - mean * mean;
    const float rstd = rsqrtf(var + 1e-5f);
#pragma unroll
    for (int j = 0; j < 6; ++j) {
        const int d = lane + 32 * j;
        const float v = (x[j] - mean) * rstd * g[d] + bta[d];
        outF[tok * cD + d] = v;
        outB[tok * cD + d] = (__bf16)v;
    }
}

// ---------------------------------------------------------------------------
// Fused attention per (batch, head): scores = Q K^T / sqrt(32), softmax,
// ctx = P V.  S=60 padded to 64.  4 waves; wave w owns query rows 16w..16w+15.
// ---------------------------------------------------------------------------
__global__ __launch_bounds__(128)
void attn_kernel(const __bf16* __restrict__ qkv, __bf16* __restrict__ ctx) {
    __shared__ __align__(32) __bf16 Pl[64 * 64];   // probabilities (bf16)
    __shared__ __align__(32) __bf16 VT[32 * 64];   // V transposed [d][s]

    const int tid  = threadIdx.x;
    const int lane = tid & 31;
    const int wave = tid >> 5;
    const int l16  = lane & 15;
    const int half = lane >> 4;
    const int b = blockIdx.x / cH;
    const int h = blockIdx.x % cH;

    const __bf16* base = qkv + (size_t)b * cS * 576;
    const int qoff = h * 32, koff = cD + h * 32, voff = 2 * cD + h * 32;

    // stage V^T into LDS (pad s>=60 with zeros)
    for (int i = tid; i < 32 * 64; i += 128) {
        const int d = i >> 6, s = i & 63;
        VT[i] = (s < cS) ? base[(size_t)s * 576 + voff + d] : (__bf16)0.0f;
    }

    // Q fragment for this wave's 16 query rows
    const int r = wave * 16 + l16;
    v16bf aq = {};
    if (r < cS) {
        const __bf16* qp = base + (size_t)r * 576 + qoff;
        v8bf lo = *(const v8bf*)(qp + 16 * half);
        v8bf hi = *(const v8bf*)(qp + 16 * half + 8);
#pragma unroll
        for (int i = 0; i < 8; ++i) { aq[i] = lo[i]; aq[i + 8] = hi[i]; }
    }

    // scores: 4 column tiles of 16; B = K^T so lane col c = key index
    v8f sc[4];
#pragma unroll
    for (int t = 0; t < 4; ++t) {
        const int c = t * 16 + l16;
        v16bf bk = {};
        if (c < cS) bk = *(const v16bf*)(base + (size_t)c * 576 + koff + 16 * half);
        v8f z = {};
        sc[t] = __builtin_amdgcn_wmma_f32_16x16x32_bf16(false, aq, false, bk,
                                                        (short)0, z, false, false);
    }

    // softmax over key dim (row lives in one 16-lane half; xor masks 1,2,4,8)
    const float scale = 0.17677669529663687f;  // 1/sqrt(32)
    float ps[4][8];
#pragma unroll
    for (int t = 0; t < 4; ++t) {
        const int c = t * 16 + l16;
#pragma unroll
        for (int v = 0; v < 8; ++v)
            ps[t][v] = (c < cS) ? sc[t][v] * scale : -3.0e38f;
    }
#pragma unroll
    for (int v = 0; v < 8; ++v) {
        float mx = fmaxf(fmaxf(ps[0][v], ps[1][v]), fmaxf(ps[2][v], ps[3][v]));
#pragma unroll
        for (int m = 1; m < 16; m <<= 1) mx = fmaxf(mx, __shfl_xor(mx, m, 32));
        float sum = 0.0f;
#pragma unroll
        for (int t = 0; t < 4; ++t) {
            ps[t][v] = __expf(ps[t][v] - mx);
            sum += ps[t][v];
        }
#pragma unroll
        for (int m = 1; m < 16; m <<= 1) sum += __shfl_xor(sum, m, 32);
        const float inv = 1.0f / sum;
        const int rr = wave * 16 + 8 * half + v;   // C-layout row
#pragma unroll
        for (int t = 0; t < 4; ++t)
            Pl[(size_t)rr * 64 + t * 16 + l16] = (__bf16)(ps[t][v] * inv);
    }
    __syncthreads();

    // ctx = P[64x64] @ V[64x32] via 2 N-tiles x 2 K-steps
    const __bf16* prow = Pl + (size_t)(wave * 16 + l16) * 64;
#pragma unroll
    for (int t2 = 0; t2 < 2; ++t2) {
        const __bf16* vrow = VT + (size_t)(t2 * 16 + l16) * 64;
        v8f cacc = {};
#pragma unroll
        for (int k0 = 0; k0 < 64; k0 += 32) {
            v8bf lo = *(const v8bf*)(prow + k0 + 16 * half);
            v8bf hi = *(const v8bf*)(prow + k0 + 16 * half + 8);
            v16bf a;
#pragma unroll
            for (int i = 0; i < 8; ++i) { a[i] = lo[i]; a[i + 8] = hi[i]; }
            v16bf bv = *(const v16bf*)(vrow + k0 + 16 * half);
            cacc = __builtin_amdgcn_wmma_f32_16x16x32_bf16(false, a, false, bv,
                                                           (short)0, cacc, false, false);
        }
        const int rbase = wave * 16 + 8 * half;
#pragma unroll
        for (int v = 0; v < 8; ++v) {
            const int rq = rbase + v;
            if (rq < cS)
                ctx[((size_t)b * cS + rq) * cD + h * 32 + t2 * 16 + l16] = (__bf16)cacc[v];
        }
    }
}

// ---------------------------------------------------------------------------
// Host orchestration
// ---------------------------------------------------------------------------
extern "C" void kernel_launch(void* const* d_in, const int* in_sizes, int n_in,
                              void* d_out, int out_size, void* d_ws, size_t ws_size,
                              hipStream_t stream) {
    const float* x    = (const float*)d_in[0];
    const float* W_in = (const float*)d_in[1];
    const float* b_in = (const float*)d_in[2];
    const float* pe   = (const float*)d_in[3];
    const float* Wq   = (const float*)d_in[4];
    const float* bq   = (const float*)d_in[5];
    const float* Wk   = (const float*)d_in[6];
    const float* bk   = (const float*)d_in[7];
    const float* Wv   = (const float*)d_in[8];
    const float* bv   = (const float*)d_in[9];
    const float* Wo   = (const float*)d_in[10];
    const float* bo   = (const float*)d_in[11];
    const float* W1   = (const float*)d_in[12];
    const float* b1   = (const float*)d_in[13];
    const float* W2   = (const float*)d_in[14];
    const float* b2   = (const float*)d_in[15];
    const float* g1   = (const float*)d_in[16];
    const float* be1  = (const float*)d_in[17];
    const float* g2   = (const float*)d_in[18];
    const float* be2  = (const float*)d_in[19];
    const float* Wc1  = (const float*)d_in[20];
    const float* bc1  = (const float*)d_in[21];
    const float* Wc2  = (const float*)d_in[22];
    const float* bc2  = (const float*)d_in[23];
    float* out = (float*)d_out;

    // ---- workspace carve-up ----
    char* wsb = (char*)d_ws;
    size_t off = 0;
    auto take = [&](size_t bytes) -> void* {
        size_t a = (off + 255) & ~(size_t)255;
        off = a + bytes;
        return (void*)(wsb + a);
    };
    float*  hf32 = (float*)take((size_t)cM * cD * 4);   // residual stream (fp32)
    float*  tmpf = (float*)take((size_t)cM * cD * 4);   // pre-LN sum (fp32)
    __bf16* hbf  = (__bf16*)take((size_t)cM * cD * 2);  // bf16 GEMM operand
    __bf16* ctxb = (__bf16*)take((size_t)cM * cD * 2);  // attention context
    // shared transient region: xpad / qkv / ffn-hidden / cls-hidden (disjoint liveness)
    __bf16* big  = (__bf16*)take((size_t)cM * cDFF * 2);
    __bf16* xbf  = big;
    __bf16* qkvb = big;   // [M][576] : q | k | v
    __bf16* ff1  = big;   // [M][768]
    __bf16* cls1 = big;   // [M][128] (96 real + pad)

    __bf16* WTin = (__bf16*)take((size_t)cD * cINP * 2);
    __bf16 *WTq[cL], *WTk[cL], *WTv[cL], *WTo[cL], *WT1[cL], *WT2[cL];
    for (int l = 0; l < cL; ++l) {
        WTq[l] = (__bf16*)take((size_t)cD * cD * 2);
        WTk[l] = (__bf16*)take((size_t)cD * cD * 2);
        WTv[l] = (__bf16*)take((size_t)cD * cD * 2);
        WTo[l] = (__bf16*)take((size_t)cD * cD * 2);
        WT1[l] = (__bf16*)take((size_t)cDFF * cD * 2);
        WT2[l] = (__bf16*)take((size_t)cD * cDFF * 2);
    }
    __bf16* WTc1 = (__bf16*)take((size_t)cDHP * cD * 2);
    __bf16* WTc2 = (__bf16*)take((size_t)cNCP * cDH * 2);
    (void)ws_size; (void)n_in; (void)in_sizes; (void)out_size;

    auto convT = [&](const float* W, __bf16* WT, int K, int Kpad, int N, int Npad) {
        int total = Npad * Kpad;
        convT_kernel<<<(total + 255) / 256, 256, 0, stream>>>(W, WT, K, Kpad, N, Npad);
    };

    // ---- weight conversion / transpose ----
    convT(W_in, WTin, cIN, cINP, cD, cD);
    for (int l = 0; l < cL; ++l) {
        convT(Wq + (size_t)l * cD * cD,  WTq[l], cD, cD, cD, cD);
        convT(Wk + (size_t)l * cD * cD,  WTk[l], cD, cD, cD, cD);
        convT(Wv + (size_t)l * cD * cD,  WTv[l], cD, cD, cD, cD);
        convT(Wo + (size_t)l * cD * cD,  WTo[l], cD, cD, cD, cD);
        convT(W1 + (size_t)l * cD * cDFF, WT1[l], cD, cD, cDFF, cDFF);
        convT(W2 + (size_t)l * cDFF * cD, WT2[l], cDFF, cDFF, cD, cD);
    }
    convT(Wc1, WTc1, cD, cD, cDH, cDHP);
    convT(Wc2, WTc2, cDH, cDH, cNC, cNCP);

    // ---- input projection + positional encoding ----
    {
        size_t total = (size_t)cM * cINP;
        padx_kernel<<<(int)((total + 255) / 256), 256, 0, stream>>>(x, xbf);
    }
    const int MB = cM / 128;  // 960 blocks of 128 rows
    gemm_wmma_bf16<1, cINP><<<dim3(MB, cD / 64), 256, 0, stream>>>(
        xbf, cINP, WTin, cINP, b_in, pe, hf32, hbf, cD, cD);

    // ---- encoder layers ----
    for (int l = 0; l < cL; ++l) {
        gemm_wmma_bf16<0, cD><<<dim3(MB, cD / 64), 256, 0, stream>>>(
            hbf, cD, WTq[l], cD, bq + (size_t)l * cD, nullptr, nullptr,
            qkvb + 0, 576, cD);
        gemm_wmma_bf16<0, cD><<<dim3(MB, cD / 64), 256, 0, stream>>>(
            hbf, cD, WTk[l], cD, bk + (size_t)l * cD, nullptr, nullptr,
            qkvb + cD, 576, cD);
        gemm_wmma_bf16<0, cD><<<dim3(MB, cD / 64), 256, 0, stream>>>(
            hbf, cD, WTv[l], cD, bv + (size_t)l * cD, nullptr, nullptr,
            qkvb + 2 * cD, 576, cD);

        attn_kernel<<<cB * cH, 128, 0, stream>>>(qkvb, ctxb);

        gemm_wmma_bf16<2, cD><<<dim3(MB, cD / 64), 256, 0, stream>>>(
            ctxb, cD, WTo[l], cD, bo + (size_t)l * cD, hf32, tmpf, nullptr,
            cD, cD);
        ln_kernel<<<cM / 8, 256, 0, stream>>>(
            tmpf, g1 + (size_t)l * cD, be1 + (size_t)l * cD, hf32, hbf);

        gemm_wmma_bf16<3, cD><<<dim3(MB, cDFF / 64), 256, 0, stream>>>(
            hbf, cD, WT1[l], cD, b1 + (size_t)l * cDFF, nullptr, nullptr, ff1,
            cDFF, cDFF);
        gemm_wmma_bf16<2, cDFF><<<dim3(MB, cD / 64), 256, 0, stream>>>(
            ff1, cDFF, WT2[l], cDFF, b2 + (size_t)l * cD, hf32, tmpf, nullptr,
            cD, cD);
        ln_kernel<<<cM / 8, 256, 0, stream>>>(
            tmpf, g2 + (size_t)l * cD, be2 + (size_t)l * cD, hf32, hbf);
    }

    // ---- classifier head ----
    gemm_wmma_bf16<3, cD><<<dim3(MB, cDHP / 64), 256, 0, stream>>>(
        hbf, cD, WTc1, cD, bc1, nullptr, nullptr, cls1, cDHP, cDH);
    gemm_wmma_bf16<4, cDH><<<dim3(MB, cNCP / 64), 256, 0, stream>>>(
        cls1, cDHP, WTc2, cDH, bc2, nullptr, out, nullptr, cNC, cNC);
}
